// ChiStream_58669253263563
// MI455X (gfx1250) — compile-verified, matrix operand order checked by reading, not run on hardware
//
#include <hip/hip_runtime.h>

// ---------------------------------------------------------------------------
// Types for CDNA5 WMMA (wave32): v_wmma_f32_16x16x32_bf16
// ---------------------------------------------------------------------------
typedef __bf16 bf16_t;
typedef bf16_t bf16x4 __attribute__((ext_vector_type(4)));
typedef bf16_t bf16x8 __attribute__((ext_vector_type(8)));
typedef bf16_t v16bf  __attribute__((ext_vector_type(16)));
typedef float  v8f    __attribute__((ext_vector_type(8)));

__device__ __forceinline__ v8f wmma_bf16(v16bf a, v16bf b, v8f c) {
  // (neg_a, A, neg_b, B, c_mod, C, reuse_a, reuse_b)
  return __builtin_amdgcn_wmma_f32_16x16x32_bf16(false, a, false, b, (short)0, c,
                                                 false, false);
}

// Load one 16-bit A/B fragment (16xK=32 or K=32x16 tile row) from an LDS tile
// stored as [row][k] with 32 bf16 per row.  Per ISA 16-bit A layout:
//  lane<16 : K runs {0..7, 16..23}; lane>=16 : K runs {8..15, 24..31}.
__device__ __forceinline__ v16bf ld_frag(const bf16_t* rowp, int lane) {
  int koff = (lane & 16) ? 8 : 0;
  bf16x8 lo = *(const bf16x8*)(rowp + koff);
  bf16x8 hi = *(const bf16x8*)(rowp + koff + 16);
  return __builtin_shufflevector(lo, hi, 0, 1, 2, 3, 4, 5, 6, 7,
                                 8, 9, 10, 11, 12, 13, 14, 15);
}

// ---------------------------------------------------------------------------
// Weight prep: W[k][n] f32  ->  Wt[n][k] bf16   (512x512)
// ---------------------------------------------------------------------------
__global__ __launch_bounds__(256) void wconv_kernel(const float* __restrict__ w,
                                                    bf16_t* __restrict__ wt) {
  int i = blockIdx.x * 256 + threadIdx.x;  // 262144 elements
  int n = i >> 9, k = i & 511;
  wt[i] = (bf16_t)w[k * 512 + n];
}

// ---------------------------------------------------------------------------
// Pack original layout (B=4,H=8,G,N,C=64) -> merged bf16 rows (b,g,n)x512,
// optional concat along N (src0 has N0 rows, src1 has N1).
// ---------------------------------------------------------------------------
__global__ __launch_bounds__(256) void pack_catN_kernel(
    bf16_t* __restrict__ dst, const float* __restrict__ s0,
    const float* __restrict__ s1, int G, int N0, int N1) {
  int Nt = N0 + N1;
  long long i = (long long)blockIdx.x * 256 + threadIdx.x;
  long long rows = (long long)4 * G * Nt;
  if (i >= rows * 128) return;
  long long t = i >> 7;
  int qq = (int)(i & 127);
  int h = qq >> 4, c4 = (qq & 15) * 4;
  int b = (int)(t / ((long long)G * Nt));
  int rem = (int)(t % ((long long)G * Nt));
  int g = rem / Nt, n = rem % Nt;
  const float* src; int nn, Ns;
  if (n < N0) { src = s0; nn = n; Ns = N0; }
  else        { src = s1; nn = n - N0; Ns = N1; }
  float4 f = *(const float4*)(src +
      ((((long long)(b * 8 + h) * G + g) * Ns + nn) * 64 + c4));
  bf16x4 r = {(bf16_t)f.x, (bf16_t)f.y, (bf16_t)f.z, (bf16_t)f.w};
  *(bf16x4*)(dst + t * 512 + qq * 4) = r;
}

// Same but concat along G: src0 has G0 groups, src1 has Gt-G0.
__global__ __launch_bounds__(256) void pack_catG_kernel(
    bf16_t* __restrict__ dst, const float* __restrict__ s0,
    const float* __restrict__ s1, int Gt, int G0, int N) {
  long long i = (long long)blockIdx.x * 256 + threadIdx.x;
  long long rows = (long long)4 * Gt * N;
  if (i >= rows * 128) return;
  long long t = i >> 7;
  int qq = (int)(i & 127);
  int h = qq >> 4, c4 = (qq & 15) * 4;
  int b = (int)(t / ((long long)Gt * N));
  int rem = (int)(t % ((long long)Gt * N));
  int g = rem / N, n = rem % N;
  const float* src; int gg, Gs;
  if (g < G0) { src = s0; gg = g; Gs = G0; }
  else        { src = s1; gg = g - G0; Gs = Gt - G0; }
  float4 f = *(const float4*)(src +
      ((((long long)(b * 8 + h) * Gs + gg) * N + n) * 64 + c4));
  bf16x4 r = {(bf16_t)f.x, (bf16_t)f.y, (bf16_t)f.z, (bf16_t)f.w};
  *(bf16x4*)(dst + t * 512 + qq * 4) = r;
}

// Concat two merged bf16 buffers into merged bf16 (mode 0: split on N, 1: on G)
__global__ __launch_bounds__(256) void concat_rows_kernel(
    bf16_t* __restrict__ dst, const bf16_t* __restrict__ s0,
    const bf16_t* __restrict__ s1, int Gt, int Nt, int mode, int split) {
  long long i = (long long)blockIdx.x * 256 + threadIdx.x;
  long long rows = (long long)4 * Gt * Nt;
  if (i >= rows * 64) return;
  long long t = i >> 6;
  int qq = (int)(i & 63);
  int b = (int)(t / ((long long)Gt * Nt));
  int rem = (int)(t % ((long long)Gt * Nt));
  int g = rem / Nt, n = rem % Nt;
  const bf16_t* src; long long srow;
  if (mode == 0) {
    if (n < split) { src = s0; srow = (long long)(b * Gt + g) * split + n; }
    else { src = s1; srow = (long long)(b * Gt + g) * (Nt - split) + (n - split); }
  } else {
    if (g < split) { src = s0; srow = (long long)(b * split + g) * Nt + n; }
    else { src = s1; srow = (long long)(b * (Gt - split) + (g - split)) * Nt + n; }
  }
  *(bf16x8*)(dst + t * 512 + qq * 8) = *(const bf16x8*)(src + srow * 512 + qq * 8);
}

// ---------------------------------------------------------------------------
// GEMM:  Y[T x 512] f32 = A[T x 512] bf16  x  W[512 x 512]  + bias
// Wt is the pre-transposed bf16 weight: Wt[n][k].
// Block = 256 thr (8 waves), block tile 128(M) x 128(N), wave tile 64x32.
// ---------------------------------------------------------------------------
__global__ __launch_bounds__(256) void gemm_bf16_kernel(
    const bf16_t* __restrict__ A, const bf16_t* __restrict__ Wt,
    const float* __restrict__ bias, float* __restrict__ Y, int T) {
  __shared__ __align__(32) bf16_t lA[128 * 32];  // [m][k]
  __shared__ __align__(32) bf16_t lB[128 * 32];  // [n][k]
  const int tid  = threadIdx.x;
  const int lane = tid & 31;
  const int wave = tid >> 5;
  const int wm = wave >> 2;  // 0..1 : which 64-row M half
  const int wn = wave & 3;   // 0..3 : which 32-col N quarter
  const int m0 = blockIdx.x * 128;
  const int n0 = blockIdx.y * 128;

  v8f acc[4][2];
#pragma unroll
  for (int i = 0; i < 4; i++)
#pragma unroll
    for (int j = 0; j < 2; j++) acc[i][j] = (v8f)0.f;

  const int sr = tid >> 1;            // staging row 0..127
  const int sc = (tid & 1) * 16;      // staging k-offset 0/16

  for (int k0 = 0; k0 < 512; k0 += 32) {
    __syncthreads();
    // Stage A tile 128x32 and B tile 128(n)x32(k) into LDS (b128 copies).
    {
      const bf16x8* sa = (const bf16x8*)(A + (long long)(m0 + sr) * 512 + k0 + sc);
      bf16x8* da = (bf16x8*)(lA + sr * 32 + sc);
      da[0] = sa[0]; da[1] = sa[1];
      const bf16x8* sb = (const bf16x8*)(Wt + (long long)(n0 + sr) * 512 + k0 + sc);
      bf16x8* db = (bf16x8*)(lB + sr * 32 + sc);
      db[0] = sb[0]; db[1] = sb[1];
    }
    if (k0 + 32 < 512) {  // prefetch next K slice (global_prefetch_b8)
      __builtin_prefetch(A + (long long)(m0 + sr) * 512 + k0 + 32 + sc, 0, 1);
      __builtin_prefetch(Wt + (long long)(n0 + sr) * 512 + k0 + 32 + sc, 0, 1);
    }
    __syncthreads();

    v16bf bfrag[2];
#pragma unroll
    for (int j = 0; j < 2; j++)
      bfrag[j] = ld_frag(lB + (wn * 32 + j * 16 + (lane & 15)) * 32, lane);
    v16bf afrag[4];
#pragma unroll
    for (int i = 0; i < 4; i++)
      afrag[i] = ld_frag(lA + (wm * 64 + i * 16 + (lane & 15)) * 32, lane);
#pragma unroll
    for (int j = 0; j < 2; j++)
#pragma unroll
      for (int i = 0; i < 4; i++)
        acc[i][j] = wmma_bf16(afrag[i], bfrag[j], acc[i][j]);
  }

  // Epilogue per ISA C layout: VGPR v, lanes0-15 -> M=v ; lanes16-31 -> M=v+8.
  const int mhi = (lane & 16) ? 8 : 0;
#pragma unroll
  for (int j = 0; j < 2; j++) {
    int n = n0 + wn * 32 + j * 16 + (lane & 15);
    float bv = bias[n];
#pragma unroll
    for (int i = 0; i < 4; i++) {
      int mb = m0 + wm * 64 + i * 16 + mhi;
#pragma unroll
      for (int v = 0; v < 8; v++)
        Y[(long long)(mb + v) * 512 + n] = acc[i][j][v] + bv;
    }
  }
}

// ---------------------------------------------------------------------------
// Flash attention, one wave per query row.  Groups decomposed (e0,e1,e2),
// per-tensor offsets in floats; each lane owns channels {2*lane, 2*lane+1}.
// ---------------------------------------------------------------------------
__global__ __launch_bounds__(256) void attn_kernel(
    const float* __restrict__ Q, const float* __restrict__ K,
    const float* __restrict__ V, float* __restrict__ O,
    int e0, int e1, int e2,
    long long qs0, long long qs1, long long qs2, long long qsr,
    long long ks0, long long ks1, long long ks2, long long ksr,
    long long os0, long long os1, long long os2, long long osr,
    int q_len, int k_len, float scale) {
  long long w = (long long)blockIdx.x * 8 + (threadIdx.x >> 5);
  long long total = (long long)e0 * e1 * e2 * q_len;
  if (w >= total) return;
  int lane = threadIdx.x & 31;
  int i = (int)(w % q_len);
  long long grp = w / q_len;
  int u2 = (int)(grp % e2);
  long long g01 = grp / e2;
  int u1 = (int)(g01 % e1);
  int u0 = (int)(g01 / e1);

  long long qoff = (long long)u0 * qs0 + (long long)u1 * qs1 +
                   (long long)u2 * qs2 + (long long)i * qsr;
  long long kbase = (long long)u0 * ks0 + (long long)u1 * ks1 + (long long)u2 * ks2;
  long long obase = (long long)u0 * os0 + (long long)u1 * os1 +
                    (long long)u2 * os2 + (long long)i * osr;

  float2 q = *(const float2*)(Q + qoff + lane * 2);
  float m = -1e30f, l = 0.f;
  float2 o; o.x = 0.f; o.y = 0.f;
  for (int j = 0; j < k_len; j++) {
    const float* kr = K + kbase + (long long)j * ksr;
    float2 kv = *(const float2*)(kr + lane * 2);
    float s = q.x * kv.x + q.y * kv.y;
#pragma unroll
    for (int off = 16; off; off >>= 1) s += __shfl_xor(s, off);
    s *= scale;
    float mn = fmaxf(m, s);
    float corr = __expf(m - mn);
    float p = __expf(s - mn);
    l = l * corr + p;
    const float* vr = V + kbase + (long long)j * ksr;
    float2 vv = *(const float2*)(vr + lane * 2);
    o.x = o.x * corr + p * vv.x;
    o.y = o.y * corr + p * vv.y;
    m = mn;
  }
  float invl = 1.f / l;
  float2 r; r.x = o.x * invl; r.y = o.y * invl;
  *(float2*)(O + obase + lane * 2) = r;
}

// ---------------------------------------------------------------------------
// LayerNorm over 512 channels, one wave per row.  Optional f32 (may alias
// input) and/or bf16 outputs.
// ---------------------------------------------------------------------------
__global__ __launch_bounds__(256) void ln_kernel(
    const float* __restrict__ x, const float* __restrict__ gamma,
    const float* __restrict__ beta, float* __restrict__ outf,
    bf16_t* __restrict__ outb, int rows) {
  int row = blockIdx.x * 8 + (threadIdx.x >> 5);
  if (row >= rows) return;
  int lane = threadIdx.x & 31;
  const float* r = x + (long long)row * 512;
  float vals[16], s = 0.f, s2 = 0.f;
#pragma unroll
  for (int i = 0; i < 16; i++) {
    float v = r[lane + i * 32];
    vals[i] = v; s += v; s2 += v * v;
  }
#pragma unroll
  for (int off = 16; off; off >>= 1) {
    s += __shfl_xor(s, off);
    s2 += __shfl_xor(s2, off);
  }
  float mu = s * (1.f / 512.f);
  float var = s2 * (1.f / 512.f) - mu * mu;
  float inv = rsqrtf(var + 1e-5f);
#pragma unroll
  for (int i = 0; i < 16; i++) {
    int c = lane + i * 32;
    float y = (vals[i] - mu) * inv * gamma[c] + beta[c];
    if (outf) outf[(long long)row * 512 + c] = y;
    if (outb) outb[(long long)row * 512 + c] = (bf16_t)y;
  }
}

// ---------------------------------------------------------------------------
// Final slices: merged f32 (b,g,n)x512 -> original (B,H,G,N,C) y/q outputs.
// ---------------------------------------------------------------------------
__global__ __launch_bounds__(256) void slice1_kernel(const float* __restrict__ L,
                                                     float* __restrict__ out) {
  // rows (b, g in 4, n in 1280); y1 = n<1024, q1 = rest
  long long i = (long long)blockIdx.x * 256 + threadIdx.x;
  if (i >= 20480LL * 128) return;
  long long t = i >> 7;
  int qq = (int)(i & 127);
  int h = qq >> 4, c4 = (qq & 15) * 4;
  int b = (int)(t / 5120);
  int rem = (int)(t % 5120);
  int g = rem / 1280, n = rem % 1280;
  float4 v = *(const float4*)(L + t * 512 + qq * 4);
  float* dst;
  if (n < 1024)
    dst = out + ((((long long)(b * 8 + h) * 4 + g) * 1024 + n) * 64 + c4);
  else
    dst = out + 8388608LL +
          ((((long long)(b * 8 + h) * 4 + g) * 256 + (n - 1024)) * 64 + c4);
  *(float4*)dst = v;
}

__global__ __launch_bounds__(256) void slice2_kernel(const float* __restrict__ L,
                                                     float* __restrict__ out) {
  // rows (b, g in 20, n in 256); y2 = g<16, q2 = rest
  long long i = (long long)blockIdx.x * 256 + threadIdx.x;
  if (i >= 20480LL * 128) return;
  long long t = i >> 7;
  int qq = (int)(i & 127);
  int h = qq >> 4, c4 = (qq & 15) * 4;
  int b = (int)(t / 5120);
  int rem = (int)(t % 5120);
  int g = rem / 256, n = rem % 256;
  float4 v = *(const float4*)(L + t * 512 + qq * 4);
  float* dst;
  if (g < 16)
    dst = out + 10485760LL +
          ((((long long)(b * 8 + h) * 16 + g) * 256 + n) * 64 + c4);
  else
    dst = out + 18874368LL +
          ((((long long)(b * 8 + h) * 4 + (g - 16)) * 256 + n) * 64 + c4);
  *(float4*)dst = v;
}

// ---------------------------------------------------------------------------
// Host orchestration
// ---------------------------------------------------------------------------
// Param leaf order: 4 inputs, then params flattened with sorted dict keys.
enum {
  iK11b = 4,  iK11w, iK12b, iK12w, iK13b, iK13w,
  iK21b = 10, iK21w, iK22b, iK22w, iK23b, iK23w,
  iQ11b = 16, iQ11w, iQ12b, iQ12w, iQ13b, iQ13w,
  iQ21b = 22, iQ21w, iQ22b, iQ22w, iQ23b, iQ23w,
  iV11b = 28, iV11w, iV12b, iV12w, iV13b, iV13w,
  iV21b = 34, iV21w, iV22b, iV22w, iV23b, iV23w,
  in11b = 40, in11g, in12b, in12g, in13b, in13g,
  in21b = 46, in21g, in22b, in22g, in23b, in23g
};

extern "C" void kernel_launch(void* const* d_in, const int* in_sizes, int n_in,
                              void* d_out, int out_size, void* d_ws,
                              size_t ws_size, hipStream_t stream) {
  (void)in_sizes; (void)n_in; (void)out_size; (void)ws_size;
  const float* x1 = (const float*)d_in[0];
  const float* p1 = (const float*)d_in[1];
  const float* x2 = (const float*)d_in[2];
  const float* p2 = (const float*)d_in[3];
  auto F = [&](int i) { return (const float*)d_in[i]; };

  char* ws = (char*)d_ws;
  size_t off = 0;
  auto carve = [&](size_t bytes) -> char* {
    char* p = ws + off;
    off = (off + bytes + 255) & ~(size_t)255;
    return p;
  };
  bf16_t* WB   = (bf16_t*)carve(18ull * 512 * 512 * 2);
  bf16_t* Xc1  = (bf16_t*)carve(20480ull * 512 * 2);
  bf16_t* Xc2  = (bf16_t*)carve(20480ull * 512 * 2);
  bf16_t* Ap1  = (bf16_t*)carve(4096ull * 512 * 2);
  bf16_t* Ap2  = (bf16_t*)carve(4096ull * 512 * 2);
  bf16_t* X1h  = (bf16_t*)carve(4096ull * 512 * 2);
  bf16_t* X2h  = (bf16_t*)carve(4096ull * 512 * 2);
  bf16_t* X12h = (bf16_t*)carve(8192ull * 512 * 2);
  bf16_t* X21h = (bf16_t*)carve(8192ull * 512 * 2);
  bf16_t* X12  = (bf16_t*)carve(20480ull * 512 * 2);
  bf16_t* X21  = (bf16_t*)carve(20480ull * 512 * 2);
  bf16_t* X12f = (bf16_t*)carve(40960ull * 512 * 2);
  bf16_t* X21f = (bf16_t*)carve(40960ull * 512 * 2);
  float*  QS   = (float*)carve(20480ull * 512 * 4);
  float*  KS   = (float*)carve(40960ull * 512 * 4);
  float*  VS   = (float*)carve(40960ull * 512 * 4);
  float*  OS   = (float*)carve(20480ull * 512 * 4);
  float*  out  = (float*)d_out;

  // Weight transpose+convert (18 GEMM weights).
  for (int widx = 5; widx <= 39; widx += 2)
    wconv_kernel<<<1024, 256, 0, stream>>>(F(widx),
                                           WB + (size_t)((widx - 5) / 2) * 262144);

  auto gemm = [&](const bf16_t* Abuf, int widx, int bidx, float* outp, int T) {
    gemm_bf16_kernel<<<dim3(T / 128, 4), 256, 0, stream>>>(
        Abuf, WB + (size_t)((widx - 5) / 2) * 262144, F(bidx), outp, T);
  };
  auto ln = [&](const float* xp, int gidx, int bidx, float* of, bf16_t* ob,
                int rows) {
    ln_kernel<<<rows / 8, 256, 0, stream>>>(xp, F(gidx), F(bidx), of, ob, rows);
  };
  auto attn = [&](int e0, int e1, int e2, long long qs0, long long qs1,
                  long long qs2, long long qsr, long long ks0, long long ks1,
                  long long ks2, long long ksr, long long os0, long long os1,
                  long long os2, long long osr, int q_len, int k_len) {
    long long waves = (long long)e0 * e1 * e2 * q_len;
    attn_kernel<<<(unsigned)((waves + 7) / 8), 256, 0, stream>>>(
        QS, KS, VS, OS, e0, e1, e2, qs0, qs1, qs2, qsr, ks0, ks1, ks2, ksr,
        os0, os1, os2, osr, q_len, k_len, 0.125f);
  };

  // ---- packs -------------------------------------------------------------
  pack_catN_kernel<<<10240, 256, 0, stream>>>(Xc1, x1, p1, 4, 1024, 256);
  pack_catG_kernel<<<10240, 256, 0, stream>>>(Xc2, x2, p2, 20, 16, 256);
  pack_catN_kernel<<<2048, 256, 0, stream>>>(Ap1, p1, p1, 4, 256, 0);
  pack_catN_kernel<<<2048, 256, 0, stream>>>(Ap2, p2, p2, 4, 256, 0);

  // ---- stage 1, stream 1 -------------------------------------------------
  gemm(Ap1, iQ11w, iQ11b, QS, 4096);
  gemm(Xc1, iK11w, iK11b, KS, 20480);
  gemm(Xc1, iV11w, iV11b, VS, 20480);
  attn(4, 4, 8, 524288, 131072, 64, 512, 2621440, 655360, 64, 512,
       524288, 131072, 64, 512, 256, 1280);
  ln(OS, in11g, in11b, nullptr, X1h, 4096);

  // ---- stage 1, stream 2 -------------------------------------------------
  gemm(Ap2, iQ21w, iQ21b, QS, 4096);
  gemm(Xc2, iK21w, iK21b, KS, 20480);
  gemm(Xc2, iV21w, iV21b, VS, 20480);
  attn(4, 8, 256, 524288, 64, 512, 131072, 2621440, 64, 512, 131072,
       524288, 64, 512, 131072, 4, 20);
  ln(OS, in21g, in21b, nullptr, X2h, 4096);

  // ---- stage 2 -----------------------------------------------------------
  concat_rows_kernel<<<2048, 256, 0, stream>>>(X12h, X1h, X2h, 4, 512, 0, 256);
  concat_rows_kernel<<<2048, 256, 0, stream>>>(X21h, X2h, X1h, 8, 256, 1, 4);

  gemm(Xc1, iQ12w, iQ12b, QS, 20480);
  gemm(X12h, iK12w, iK12b, KS, 8192);
  gemm(X12h, iV12w, iV12b, VS, 8192);
  attn(4, 4, 8, 2621440, 655360, 64, 512, 1048576, 262144, 64, 512,
       2621440, 655360, 64, 512, 1280, 512);
  ln(OS, in12g, in12b, nullptr, X12, 20480);

  gemm(Xc2, iQ22w, iQ22b, QS, 20480);
  gemm(X21h, iK22w, iK22b, KS, 8192);
  gemm(X21h, iV22w, iV22b, VS, 8192);
  attn(4, 8, 256, 2621440, 64, 512, 131072, 1048576, 64, 512, 131072,
       2621440, 64, 512, 131072, 20, 8);
  ln(OS, in22g, in22b, nullptr, X21, 20480);

  // ---- stage 3 -----------------------------------------------------------
  concat_rows_kernel<<<10240, 256, 0, stream>>>(X12f, Xc1, X12, 8, 1280, 1, 4);
  concat_rows_kernel<<<10240, 256, 0, stream>>>(X21f, Xc2, X21, 20, 512, 0, 256);

  gemm(Xc1, iQ13w, iQ13b, QS, 20480);
  gemm(X12f, iK13w, iK13b, KS, 40960);
  gemm(X12f, iV13w, iV13b, VS, 40960);
  attn(4, 8, 1280, 2621440, 64, 512, 655360, 5242880, 64, 512, 655360,
       2621440, 64, 512, 655360, 4, 8);
  ln(OS, in13g, in13b, OS, nullptr, 20480);
  slice1_kernel<<<10240, 256, 0, stream>>>(OS, out);

  gemm(Xc2, iQ23w, iQ23b, QS, 20480);
  gemm(X21f, iK23w, iK23b, KS, 40960);
  gemm(X21f, iV23w, iV23b, VS, 40960);
  attn(4, 20, 8, 2621440, 131072, 64, 512, 5242880, 262144, 64, 512,
       2621440, 131072, 64, 512, 256, 512);
  ln(OS, in23g, in23b, OS, nullptr, 20480);
  slice2_kernel<<<10240, 256, 0, stream>>>(OS, out);
}
